// FeatureAttention_49684181680937
// MI455X (gfx1250) — compile-verified, matrix operand order checked by reading, not run on hardware
//
#include <hip/hip_runtime.h>
#include <hip/hip_bf16.h>
#include <math.h>

// Sizes from the reference
#define NN 8192
#define DD 512
#define UU 256

typedef __attribute__((ext_vector_type(8)))  __bf16 v8bf;
typedef __attribute__((ext_vector_type(16))) __bf16 v16bf;
typedef __attribute__((ext_vector_type(8)))  float  v8f;

// D = A(16x32 bf16) * B(32x16 bf16) + C(16x16 f32)
__device__ __forceinline__ v8f wmma_bf16(v16bf a, v16bf b, v8f c) {
  return __builtin_amdgcn_wmma_f32_16x16x32_bf16(
      /*neg_a=*/false, a, /*neg_b=*/false, b,
      /*c_mod=*/(short)0, c, /*reuse_a=*/false, /*reuse_b=*/false);
}

// Load a 16x32 bf16 A-fragment (or 32x16 B-fragment) from a row-major
// [rows][ld] bf16 matrix. Per the CDNA5 16-bit A/B layout:
//   lanes 0-15 : row = lr, K = k0+0..7 (VGPR0-3), k0+16..23 (VGPR4-7)
//   lanes 16-31: row = lr, K = k0+8..15,          k0+24..31
// -> two contiguous 16-byte loads per lane.
__device__ __forceinline__ v16bf load_frag(const __bf16* __restrict__ base,
                                           int row, int ld, int k0, int hh) {
  const __bf16* p = base + (size_t)row * (size_t)ld + k0 + hh * 8;
  union { v16bf v; v8bf h[2]; } u;
  u.h[0] = *(const v8bf*)(p);
  u.h[1] = *(const v8bf*)(p + 16);
  return u.v;
}

// Async DMA 16 bytes/lane from global to LDS (ASYNCcnt-tracked).
// VDST = LDS byte address (low 32 bits of the generic shared pointer:
// flat->LDS aperture mapping truncates to addr[31:0]).
__device__ __forceinline__ void async_g2l_b128(const __bf16* gaddr, __bf16* laddr) {
  uint32_t l = (uint32_t)(uintptr_t)laddr;
  asm volatile("global_load_async_to_lds_b128 %0, %1, off"
               :: "v"(l), "v"(gaddr) : "memory");
}

// ---------------------------------------------------------------- conversions
__global__ void cvt_bf16_kernel(const float* __restrict__ in,
                                __bf16* __restrict__ out, int n) {
  int i = blockIdx.x * blockDim.x + threadIdx.x;
  if (i < n) out[i] = (__bf16)in[i];
}

// in[R][C] f32 -> out[C][R] bf16
__global__ void transpose_cvt_kernel(const float* __restrict__ in,
                                     __bf16* __restrict__ out, int R, int C) {
  int i = blockIdx.x * blockDim.x + threadIdx.x;
  if (i < R * C) {
    int r = i / C, c = i % C;
    out[(size_t)c * R + r] = (__bf16)in[i];
  }
}

// ------------------------------------------------------------------ GEMM
// C[M][N] = A[M][K] @ Bt[N][K]^T   (A, Bt bf16 row-major)
// MODE 0: store bf16 row-major, scaled      (Q, K projections)
// MODE 1: store bf16 transposed Ct[N][M]    (V -> Vt)
// MODE 2: store f32 row-major               (final output)
// wave = 16x64 tile; block = 4 waves = 16x256; grid = (M/16, N/256)
template <int MODE>
__global__ __launch_bounds__(128) void gemm_bf16_kernel(
    const __bf16* __restrict__ A, const __bf16* __restrict__ Bt,
    void* __restrict__ Cout, int M, int N, int K, float scale) {
  const int tid  = threadIdx.x;
  const int wave = tid >> 5;
  const int lane = tid & 31;
  const int hh   = lane >> 4;   // which 16-lane half
  const int lr   = lane & 15;
  const int m0 = blockIdx.x * 16;
  const int n0 = blockIdx.y * 256 + wave * 64;

  v8f acc[4];
#pragma unroll
  for (int t = 0; t < 4; ++t) acc[t] = {};

  for (int k0 = 0; k0 < K; k0 += 32) {
    v16bf af = load_frag(A, m0 + lr, K, k0, hh);
#pragma unroll
    for (int t = 0; t < 4; ++t) {
      v16bf bfr = load_frag(Bt, n0 + t * 16 + lr, K, k0, hh);
      acc[t] = wmma_bf16(af, bfr, acc[t]);
    }
  }

  // C layout: lane lr (half hh) holds col n0+t*16+lr, rows m0 + r + 8*hh
  if (MODE == 0) {
    __bf16* C = (__bf16*)Cout;
#pragma unroll
    for (int t = 0; t < 4; ++t)
#pragma unroll
      for (int r = 0; r < 8; ++r)
        C[(size_t)(m0 + r + 8 * hh) * N + n0 + t * 16 + lr] =
            (__bf16)(acc[t][r] * scale);
  } else if (MODE == 1) {
    __bf16* C = (__bf16*)Cout;  // Ct[N][M]
#pragma unroll
    for (int t = 0; t < 4; ++t) {
      v8bf pk;
#pragma unroll
      for (int r = 0; r < 8; ++r) pk[r] = (__bf16)(acc[t][r]);
      *(v8bf*)(C + (size_t)(n0 + t * 16 + lr) * M + m0 + 8 * hh) = pk;
    }
  } else {
    float* C = (float*)Cout;
#pragma unroll
    for (int t = 0; t < 4; ++t)
#pragma unroll
      for (int r = 0; r < 8; ++r)
        C[(size_t)(m0 + r + 8 * hh) * N + n0 + t * 16 + lr] = acc[t][r];
  }
}

// ------------------------------------------------------------ flash attention
// Q, Km: bf16 [NN][UU] row-major (Q pre-scaled by 1/sqrt(U))
// Vt:    bf16 [UU][NN] (V transposed)
// O:     bf16 [NN][UU] row-major, softmax(Q Km^T) @ V
// Block = 4 waves; wave owns 16 query rows, full 256-col accumulator.
// K/V chunks are DMA'd into double-buffered LDS once per block and shared.
__global__ __launch_bounds__(128) void flash_attn_kernel(
    const __bf16* __restrict__ Q, const __bf16* __restrict__ Km,
    const __bf16* __restrict__ Vt, __bf16* __restrict__ O) {
  __shared__ __bf16 ldsK[2][32 * UU];   // [key 0..31][feat 0..255]
  __shared__ __bf16 ldsV[2][UU * 32];   // [feat 0..255][key 0..31]
  __shared__ __bf16 ldsP[4][16 * 32];   // per-wave P staging

  const int tid  = threadIdx.x;
  const int wave = tid >> 5;
  const int lane = tid & 31;
  const int hh   = lane >> 4;
  const int lr   = lane & 15;
  const int m0 = blockIdx.x * 64 + wave * 16;

  // Cooperative async stage of one 32-key chunk (16 async ops per thread,
  // i.e. 16 ASYNCcnt increments per wave).
  auto stage_chunk = [&](int kb, int buf) {
#pragma unroll
    for (int i = 0; i < 8; ++i) {           // K: 32 rows x 512B
      int e = tid + 128 * i;                // 0..1023
      int row = e >> 5;
      int seg = e & 31;                     // 16B segment in row
      async_g2l_b128(Km + (size_t)(kb + row) * UU + seg * 8,
                     &ldsK[buf][row * UU + seg * 8]);
    }
#pragma unroll
    for (int i = 0; i < 8; ++i) {           // V: 256 rows x 64B
      int e = tid + 128 * i;
      int feat = e >> 2;
      int seg  = e & 3;
      async_g2l_b128(Vt + (size_t)feat * NN + kb + seg * 8,
                     &ldsV[buf][feat * 32 + seg * 8]);
    }
  };

  // Q A-fragments resident in VGPRs (16 rows x 256 feats = 8 frags)
  v16bf qf[8];
#pragma unroll
  for (int kc = 0; kc < 8; ++kc) qf[kc] = load_frag(Q, m0 + lr, UU, kc * 32, hh);

  v8f oacc[16];
#pragma unroll
  for (int t = 0; t < 16; ++t) oacc[t] = {};
  float mrow[8], lrow[8];
#pragma unroll
  for (int r = 0; r < 8; ++r) { mrow[r] = -1e30f; lrow[r] = 0.f; }

  __bf16* myP = &ldsP[wave][0];

  stage_chunk(0, 0);   // prologue: fill buffer 0

  int buf = 0;
  for (int kb = 0; kb < NN; kb += 32, buf ^= 1) {
    // Prefetch next chunk into the other buffer, then drain only the
    // previous chunk's 16 in-flight async ops (they complete in order).
    if (kb + 32 < NN) {
      stage_chunk(kb + 32, buf ^ 1);
      asm volatile("s_wait_asynccnt 0x10" ::: "memory");
    } else {
      asm volatile("s_wait_asynccnt 0x0" ::: "memory");
    }
    __syncthreads();   // LDS tile visible to all waves

    const __bf16* Kc = &ldsK[buf][0];
    const __bf16* Vc = &ldsV[buf][0];

    // S tiles for keys kb..kb+15 and kb+16..kb+31 (ds_load fragments)
    v8f s0 = {}, s1 = {};
#pragma unroll
    for (int kc = 0; kc < 8; ++kc) {
      v16bf b0 = load_frag(Kc, lr,      UU, kc * 32, hh);
      v16bf b1 = load_frag(Kc, 16 + lr, UU, kc * 32, hh);
      s0 = wmma_bf16(qf[kc], b0, s0);
      s1 = wmma_bf16(qf[kc], b1, s1);
    }

    // online softmax; row r maps to query m0 + r + 8*hh, elements live
    // across the 16-lane half -> xor-shuffles with masks <16 stay in-half.
    float sc[8];
#pragma unroll
    for (int r = 0; r < 8; ++r) {
      float mx = fmaxf(s0[r], s1[r]);
      mx = fmaxf(mx, __shfl_xor(mx, 1, 32));
      mx = fmaxf(mx, __shfl_xor(mx, 2, 32));
      mx = fmaxf(mx, __shfl_xor(mx, 4, 32));
      mx = fmaxf(mx, __shfl_xor(mx, 8, 32));
      float mnew = fmaxf(mrow[r], mx);
      float scl  = __expf(mrow[r] - mnew);
      float p0   = __expf(s0[r] - mnew);
      float p1   = __expf(s1[r] - mnew);
      float rs = p0 + p1;
      rs += __shfl_xor(rs, 1, 32);
      rs += __shfl_xor(rs, 2, 32);
      rs += __shfl_xor(rs, 4, 32);
      rs += __shfl_xor(rs, 8, 32);
      lrow[r] = lrow[r] * scl + rs;
      mrow[r] = mnew;
      sc[r] = scl;
      s0[r] = p0;
      s1[r] = p1;
    }
#pragma unroll
    for (int t = 0; t < 16; ++t)
#pragma unroll
      for (int r = 0; r < 8; ++r) oacc[t][r] *= sc[r];

    // P: C-layout -> row-major [16][32] in LDS, then read back as A-fragment.
#pragma unroll
    for (int r = 0; r < 8; ++r) {
      myP[(r + 8 * hh) * 32 + lr]      = (__bf16)s0[r];
      myP[(r + 8 * hh) * 32 + 16 + lr] = (__bf16)s1[r];
    }
    // per-wave DS ops are in-order; drain them before the cross-lane read
    asm volatile("s_wait_dscnt 0" ::: "memory");
    v16bf pf;
    {
      union { v16bf v; v8bf h[2]; } u;
      const __bf16* pp = myP + lr * 32 + hh * 8;
      u.h[0] = *(const v8bf*)(pp);
      u.h[1] = *(const v8bf*)(pp + 16);
      pf = u.v;
    }

    // O += P @ V  (B-fragments from the LDS V tile, ld = 32)
#pragma unroll
    for (int t = 0; t < 16; ++t) {
      v16bf vf = load_frag(Vc, t * 16 + lr, 32, 0, hh);
      oacc[t] = wmma_bf16(pf, vf, oacc[t]);
    }

    __syncthreads();   // all waves done reading before next prefetch overwrites
  }

  float rinv[8];
#pragma unroll
  for (int r = 0; r < 8; ++r) rinv[r] = 1.0f / lrow[r];
#pragma unroll
  for (int t = 0; t < 16; ++t)
#pragma unroll
    for (int r = 0; r < 8; ++r)
      O[(size_t)(m0 + r + 8 * hh) * UU + t * 16 + lr] =
          (__bf16)(oacc[t][r] * rinv[r]);
}

// ------------------------------------------------------------------ launcher
extern "C" void kernel_launch(void* const* d_in, const int* in_sizes, int n_in,
                              void* d_out, int out_size, void* d_ws, size_t ws_size,
                              hipStream_t stream) {
  const float* x  = (const float*)d_in[0];
  const float* Wq = (const float*)d_in[1];
  const float* Wk = (const float*)d_in[2];
  const float* Wv = (const float*)d_in[3];
  const float* Wo = (const float*)d_in[4];
  float* out = (float*)d_out;

  char* ws = (char*)d_ws;
  __bf16* xb  = (__bf16*)ws; ws += (size_t)NN * DD * 2;
  __bf16* WqT = (__bf16*)ws; ws += (size_t)UU * DD * 2;
  __bf16* WkT = (__bf16*)ws; ws += (size_t)UU * DD * 2;
  __bf16* WvT = (__bf16*)ws; ws += (size_t)UU * DD * 2;
  __bf16* WoT = (__bf16*)ws; ws += (size_t)DD * UU * 2;
  __bf16* Qb  = (__bf16*)ws; ws += (size_t)NN * UU * 2;
  __bf16* Kb  = (__bf16*)ws; ws += (size_t)NN * UU * 2;
  __bf16* Vtb = (__bf16*)ws; ws += (size_t)UU * NN * 2;
  __bf16* Ob  = (__bf16*)ws;

  cvt_bf16_kernel<<<(NN * DD + 255) / 256, 256, 0, stream>>>(x, xb, NN * DD);
  transpose_cvt_kernel<<<(DD * UU + 255) / 256, 256, 0, stream>>>(Wq, WqT, DD, UU);
  transpose_cvt_kernel<<<(DD * UU + 255) / 256, 256, 0, stream>>>(Wk, WkT, DD, UU);
  transpose_cvt_kernel<<<(DD * UU + 255) / 256, 256, 0, stream>>>(Wv, WvT, DD, UU);
  transpose_cvt_kernel<<<(UU * DD + 255) / 256, 256, 0, stream>>>(Wo, WoT, UU, DD);

  dim3 blk(128);
  dim3 gP(NN / 16, UU / 256);
  // Q pre-scaled by 1/sqrt(U) = 1/16
  gemm_bf16_kernel<0><<<gP, blk, 0, stream>>>(xb, WqT, Qb, NN, UU, DD, 0.0625f);
  gemm_bf16_kernel<0><<<gP, blk, 0, stream>>>(xb, WkT, Kb, NN, UU, DD, 1.0f);
  gemm_bf16_kernel<1><<<gP, blk, 0, stream>>>(xb, WvT, Vtb, NN, UU, DD, 1.0f);

  flash_attn_kernel<<<dim3(NN / 64), blk, 0, stream>>>(Qb, Kb, Vtb, Ob);

  dim3 gO(NN / 16, DD / 256);
  gemm_bf16_kernel<2><<<gO, blk, 0, stream>>>(Ob, WoT, out, NN, DD, UU, 1.0f);
}
